// ActionModel2_74397423501381
// MI455X (gfx1250) — matrix operand order repaired and checked by reading, not compiled, not run on hardware
//
#include <hip/hip_runtime.h>

// ---------------- static problem dims (match reference) ----------------
#define Nn   32768     // nodes
#define Ee   524288    // edges
#define Ff   128       // node feature
#define Hh   256       // hidden
#define EDd  32        // edge feature
#define Aa   10        // actions
#define Bb   64        // graphs
#define NPGc 512       // nodes per graph

typedef __bf16 v16bf __attribute__((ext_vector_type(16)));
typedef float  v8f   __attribute__((ext_vector_type(8)));

// ---------------- bf16 helpers (RNE) ----------------
__device__ __forceinline__ unsigned short f2bfbits(float f) {
  unsigned u = __builtin_bit_cast(unsigned, f);
  u += 0x7FFFu + ((u >> 16) & 1u);
  return (unsigned short)(u >> 16);
}
__device__ __forceinline__ __bf16 f2bf(float f) {
  unsigned short h = f2bfbits(f);
  return __builtin_bit_cast(__bf16, h);
}

// Build A-fragment (16x32 bf16, CDNA5 layout):
// lane l: row = l&15, half h = l>>4; elems 0..7 -> K = h*8 + i ; 8..15 -> K = 16 + h*8 + (i-8)
// p points at row start + h*8 (floats). Optional second operand p2 added elementwise.
__device__ __forceinline__ v16bf load_a_frag(const float* __restrict__ p,
                                             const float* __restrict__ p2) {
  float4 q0 = *(const float4*)(p);
  float4 q1 = *(const float4*)(p + 4);
  float4 q2 = *(const float4*)(p + 16);
  float4 q3 = *(const float4*)(p + 20);
  if (p2) {
    float4 r0 = *(const float4*)(p2);
    float4 r1 = *(const float4*)(p2 + 4);
    float4 r2 = *(const float4*)(p2 + 16);
    float4 r3 = *(const float4*)(p2 + 20);
    q0.x += r0.x; q0.y += r0.y; q0.z += r0.z; q0.w += r0.w;
    q1.x += r1.x; q1.y += r1.y; q1.z += r1.z; q1.w += r1.w;
    q2.x += r2.x; q2.y += r2.y; q2.z += r2.z; q2.w += r2.w;
    q3.x += r3.x; q3.y += r3.y; q3.z += r3.z; q3.w += r3.w;
  }
  v16bf a;
  a[0]  = f2bf(q0.x); a[1]  = f2bf(q0.y); a[2]  = f2bf(q0.z); a[3]  = f2bf(q0.w);
  a[4]  = f2bf(q1.x); a[5]  = f2bf(q1.y); a[6]  = f2bf(q1.z); a[7]  = f2bf(q1.w);
  a[8]  = f2bf(q2.x); a[9]  = f2bf(q2.y); a[10] = f2bf(q2.z); a[11] = f2bf(q2.w);
  a[12] = f2bf(q3.x); a[13] = f2bf(q3.y); a[14] = f2bf(q3.z); a[15] = f2bf(q3.w);
  return a;
}

// ---------------- zero fill ----------------
__global__ void k_zero(float* __restrict__ p, int n) {
  for (int i = blockIdx.x * blockDim.x + threadIdx.x; i < n; i += gridDim.x * blockDim.x)
    p[i] = 0.f;
}

// ---------------- pack fp32 weight [K,Ncol] into WMMA-B bf16 tiles ----------------
// B tile = 32(K) x 16(N): lane l holds col n = l&15, half h = l>>4, elem i -> K = kt*32 + h*8 + (i<8 ? i : i+8).
// Tile storage: 512 bf16 contiguous; per-lane 16 bf16 (32B) at tile*512 + lane*16.
__global__ void k_pack(const float* __restrict__ W, unsigned short* __restrict__ P,
                       int K, int Ncol) {
  int idx = blockIdx.x * blockDim.x + threadIdx.x;
  int total = (K / 32) * (Ncol / 16) * 32;
  if (idx >= total) return;
  int lane = idx & 31, tile = idx >> 5;
  int nT = Ncol / 16;
  int kt = tile / nT, nt = tile - kt * nT;
  int n = nt * 16 + (lane & 15);
  int hl = lane >> 4;
  unsigned short* dst = P + tile * 512 + lane * 16;
#pragma unroll
  for (int i = 0; i < 16; ++i) {
    int k = kt * 32 + hl * 8 + (i < 8 ? i : i + 8);
    dst[i] = f2bfbits(W[k * Ncol + n]);
  }
}

// ---------------- fused GINE edge kernel ----------------
// Per wave: 16 edges. Packed edge-weights staged in LDS once per block.
// acc = edge_attr_tile @ We (one K=32 WMMA per 16-channel tile), then
// v = relu(acc + biasE + G[src]) and atomicAdd into agg[dst].
// Software-pipelined: next tile's WMMA issues before current epilogue.
__global__ __launch_bounds__(256) void k_edge(
    const float* __restrict__ ea, const unsigned short* __restrict__ PB,
    const float* __restrict__ biasE, const float* __restrict__ G,
    float* __restrict__ agg, const long long* __restrict__ srcI,
    const long long* __restrict__ dstI, int Cc) {
  __shared__ __align__(16) unsigned short ldsB[8192];   // up to 16 KB (Cc=256)
  int t = threadIdx.x;
  int nT = Cc >> 4;
  // cooperative stage of the whole packed weight (nT*1024 bytes)
  for (int i = t; i < nT * 64; i += 256)
    ((uint4*)ldsB)[i] = ((const uint4*)PB)[i];
  __syncthreads();

  int lane = t & 31;
  int gw = blockIdx.x * 8 + (t >> 5);
  int eBase = gw * 16;                       // exact fit: no guard, EXEC all-ones
  int hl = lane >> 4;
  int n15 = lane & 15;
  int row = eBase + n15;
  v16bf a = load_a_frag(ea + row * EDd + hl * 8, nullptr);

  int gOff[8], aOff[8];
#pragma unroll
  for (int r = 0; r < 8; ++r) {
    int e = eBase + 8 * hl + r;              // edge owned by (lane-half, reg r)
    gOff[r] = (int)srcI[e] * Cc;
    aOff[r] = (int)dstI[e] * Cc;
  }

  v8f zero = {0.f, 0.f, 0.f, 0.f, 0.f, 0.f, 0.f, 0.f};
  auto tileMM = [&](int nt) {
    v16bf b = *(const v16bf*)(ldsB + nt * 512 + lane * 16);   // ds_load_b128 x2
    return __builtin_amdgcn_wmma_f32_16x16x32_bf16(false, a, false, b,
                                                   (short)0, zero, false, false);
  };

  v8f accCur = tileMM(0);
  for (int nt = 0; nt < nT; ++nt) {
    v8f accNext = (nt + 1 < nT) ? tileMM(nt + 1) : accCur;    // hide WMMA->VALU hazard
    int n = nt * 16 + n15;
    float bv = biasE[n];
    float gv[8];
#pragma unroll
    for (int r = 0; r < 8; ++r) gv[r] = G[gOff[r] + n];       // batched gather clause
#pragma unroll
    for (int r = 0; r < 8; ++r) {
      float v = accCur[r] + bv + gv[r];
      v = fmaxf(v, 0.f);
      atomicAdd(&agg[aOff[r] + n], v);       // 16 lanes -> 16 consecutive channels
    }
    accCur = accNext;
  }
}

// ---------------- tiled bf16 WMMA GEMM: C = act((A0[+A1]) @ W + bias) ----------------
// Block = 8 waves sharing one 64-column B slice (staged in LDS); wave = 16(M) x 64(N).
// act: 0=none, 1=relu, 2=leaky_relu(0.01).  Requires M%128==0.
__global__ __launch_bounds__(256) void k_gemm(
    const float* __restrict__ A0, const float* __restrict__ A1,
    const unsigned short* __restrict__ PB, const float* __restrict__ bias,
    float* __restrict__ C, int M, int K, int Ncol, int act) {
  __shared__ __align__(16) unsigned short ldsB[16384];  // up to 32 KB (K=256)
  int t = threadIdx.x;
  int nT = Ncol >> 4;
  int kSteps = K >> 5;
  int mBlocks = M >> 7;
  int bm = blockIdx.x % mBlocks;
  int bn = blockIdx.x / mBlocks;             // shared N-slice for all 8 waves
  // stage B slice: per kt, 4 consecutive tiles (4 KB) -> LDS
  for (int kt = 0; kt < kSteps; ++kt)
    ((uint4*)(ldsB + kt * 2048))[t] = ((const uint4*)(PB + (kt * nT + bn * 4) * 512))[t];
  __syncthreads();

  int lane = t & 31;
  int hl = lane >> 4;
  int n15 = lane & 15;
  int mBase = (bm * 8 + (t >> 5)) * 16;
  int row = mBase + n15;

  v8f acc[4];
  v8f zero = {0.f, 0.f, 0.f, 0.f, 0.f, 0.f, 0.f, 0.f};
#pragma unroll
  for (int j = 0; j < 4; ++j) acc[j] = zero;

  const float* pa  = A0 + row * K + hl * 8;
  const float* pa1 = A1 ? (A1 + row * K + hl * 8) : nullptr;
  for (int kt = 0; kt < kSteps; ++kt) {
    if (kt + 1 < kSteps) {
      __builtin_prefetch(pa + (kt + 1) * 32, 0, 1);          // global_prefetch_b8
      if (pa1) __builtin_prefetch(pa1 + (kt + 1) * 32, 0, 1);
    }
    v16bf a = load_a_frag(pa + kt * 32, pa1 ? pa1 + kt * 32 : nullptr);
    const unsigned short* pb = ldsB + kt * 2048 + lane * 16;
    v16bf b0 = *(const v16bf*)(pb);
    v16bf b1 = *(const v16bf*)(pb + 512);
    v16bf b2 = *(const v16bf*)(pb + 1024);
    v16bf b3 = *(const v16bf*)(pb + 1536);
    acc[0] = __builtin_amdgcn_wmma_f32_16x16x32_bf16(false, a, false, b0, (short)0, acc[0], false, false);
    acc[1] = __builtin_amdgcn_wmma_f32_16x16x32_bf16(false, a, false, b1, (short)0, acc[1], false, false);
    acc[2] = __builtin_amdgcn_wmma_f32_16x16x32_bf16(false, a, false, b2, (short)0, acc[2], false, false);
    acc[3] = __builtin_amdgcn_wmma_f32_16x16x32_bf16(false, a, false, b3, (short)0, acc[3], false, false);
  }
#pragma unroll
  for (int j = 0; j < 4; ++j) {
    int n = (bn * 4 + j) * 16 + n15;
    float bv = bias[n];
#pragma unroll
    for (int r = 0; r < 8; ++r) {
      int m = mBase + r + 8 * hl;
      float v = acc[j][r] + bv;
      if (act == 1) v = fmaxf(v, 0.f);
      else if (act == 2) v = (v > 0.f) ? v : 0.01f * v;
      C[m * Ncol + n] = v;
    }
  }
}

// ---------------- BN column stats (sum, sumsq) over rows ----------------
__global__ void k_colstats(const float* __restrict__ Y, float* __restrict__ sums,
                           float* __restrict__ sq, int rowsPerBlock) {
  int c = threadIdx.x;                   // Hh threads
  int r0 = blockIdx.x * rowsPerBlock;
  float s = 0.f, q = 0.f;
  for (int r = 0; r < rowsPerBlock; ++r) {
    float v = Y[(r0 + r) * Hh + c];
    s += v; q += v * v;
  }
  atomicAdd(&sums[c], s);
  atomicAdd(&sq[c], q);
}

// ---------------- BN apply: act==1 -> relu ; act==2 -> sigmoid(relu(.)) ----------------
__global__ void k_bnapply(const float* __restrict__ Y, float* __restrict__ O,
                          const float* __restrict__ sums, const float* __restrict__ sq,
                          const float* __restrict__ g, const float* __restrict__ bt,
                          int total, int act, float invM) {
  for (int i = blockIdx.x * blockDim.x + threadIdx.x; i < total; i += gridDim.x * blockDim.x) {
    int c = i & (Hh - 1);
    float mean = sums[c] * invM;
    float var = sq[c] * invM - mean * mean;
    float rstd = rsqrtf(var + 1e-5f);
    float v = g[c] * (Y[i] - mean) * rstd + bt[c];
    v = fmaxf(v, 0.f);
    if (act == 2) v = 1.f / (1.f + __expf(-v));
    O[i] = v;
  }
}

// ---------------- node score GEMV: out = sigmoid(X @ w + b0) ----------------
__global__ __launch_bounds__(256) void k_nodescore(const float* __restrict__ X,
                                                   const float* __restrict__ w,
                                                   const float* __restrict__ b,
                                                   float* __restrict__ out) {
  int lane = threadIdx.x & 31;
  int row = blockIdx.x * 8 + (threadIdx.x >> 5);
  float s = 0.f;
#pragma unroll
  for (int k = lane; k < Hh; k += 32) s += X[row * Hh + k] * w[k];
#pragma unroll
  for (int off = 16; off; off >>= 1) s += __shfl_xor(s, off, 32);
  if (lane == 0) out[row] = 1.f / (1.f + __expf(-(s + b[0])));
}

// ---------------- per-graph mean pooling ----------------
__global__ void k_pool(const float* __restrict__ h2, float* __restrict__ pool) {
  int b = blockIdx.x, c = threadIdx.x;
  float s = 0.f;
  for (int r = 0; r < NPGc; ++r) s += h2[(b * NPGc + r) * Hh + c];
  pool[b * Hh + c] = s * (1.f / NPGc);
}

// ---------------- tiny row GEMM (64 x 256 x 256) ----------------
__global__ void k_rowgemm(const float* __restrict__ Ain, const float* __restrict__ W,
                          const float* __restrict__ bias, float* __restrict__ O) {
  int r = blockIdx.x, c = threadIdx.x;
  float s = bias[c];
  for (int k = 0; k < Hh; ++k) s += Ain[r * Hh + k] * W[k * Hh + c];
  O[r * Hh + c] = s;
}

// ---------------- tiny BN over 64 rows, in-place, relu ----------------
__global__ void k_smallbn(float* __restrict__ Aio, const float* __restrict__ g,
                          const float* __restrict__ bt) {
  int c = threadIdx.x;
  float s = 0.f, q = 0.f;
  for (int r = 0; r < Bb; ++r) { float v = Aio[r * Hh + c]; s += v; q += v * v; }
  float mean = s * (1.f / Bb);
  float var = q * (1.f / Bb) - mean * mean;
  float rstd = rsqrtf(var + 1e-5f);
  for (int r = 0; r < Bb; ++r) {
    float v = g[c] * (Aio[r * Hh + c] - mean) * rstd + bt[c];
    Aio[r * Hh + c] = fmaxf(v, 0.f);
  }
}

// ---------------- action head output ----------------
__global__ void k_actionout(const float* __restrict__ a2, const float* __restrict__ W,
                            const float* __restrict__ bias, float* __restrict__ out) {
  int t = threadIdx.x;
  if (t >= Bb * Aa) return;
  int r = t / Aa, j = t - r * Aa;
  float s = bias[j];
  for (int k = 0; k < Hh; ++k) s += a2[r * Hh + k] * W[k * Aa + j];
  out[t] = 1.f / (1.f + __expf(-s));
}

// =====================================================================
extern "C" void kernel_launch(void* const* d_in, const int* in_sizes, int n_in,
                              void* d_out, int out_size, void* d_ws, size_t ws_size,
                              hipStream_t stream) {
  const float*     x    = (const float*)d_in[0];
  const long long* ei   = (const long long*)d_in[1];   // [2,E] int64
  const float*     ea   = (const float*)d_in[2];
  // d_in[3] = batch (unused: uniform graphs)
  const float *We1 = (const float*)d_in[4],  *be1 = (const float*)d_in[5];
  const float *W1  = (const float*)d_in[6],  *b1  = (const float*)d_in[7];
  const float *g1  = (const float*)d_in[8],  *bt1 = (const float*)d_in[9];
  const float *We2 = (const float*)d_in[10], *be2 = (const float*)d_in[11];
  const float *W2  = (const float*)d_in[12], *b2  = (const float*)d_in[13];
  const float *g2  = (const float*)d_in[14], *bt2 = (const float*)d_in[15];
  const float *Wn1 = (const float*)d_in[16], *bn1 = (const float*)d_in[17];
  const float *Wn2 = (const float*)d_in[18], *bn2 = (const float*)d_in[19];
  const float *Wn3 = (const float*)d_in[20], *bn3 = (const float*)d_in[21];
  const float *Wa1 = (const float*)d_in[22], *ba1 = (const float*)d_in[23];
  const float *ga1 = (const float*)d_in[24], *bta1= (const float*)d_in[25];
  const float *Wa2 = (const float*)d_in[26], *ba2 = (const float*)d_in[27];
  const float *ga2 = (const float*)d_in[28], *bta2= (const float*)d_in[29];
  const float *Wa3 = (const float*)d_in[30], *ba3 = (const float*)d_in[31];

  const long long* srcI = ei;
  const long long* dstI = ei + Ee;

  float* out = (float*)d_out;                 // [0,640) action_prob, [640,640+N) node_score
  float* ws  = (float*)d_ws;

  // ---- workspace layout (float units) ----
  const size_t o_agg1 = 0;                                  // N*F
  const size_t o_agg2 = o_agg1 + (size_t)Nn * Ff;           // N*H
  const size_t o_stat = o_agg2 + (size_t)Nn * Hh;           // 1024 (s1,q1,s2,q2)
  const size_t o_y1   = o_stat + 1024;
  const size_t o_h1   = o_y1 + (size_t)Nn * Hh;
  const size_t o_y2   = o_h1 + (size_t)Nn * Hh;
  const size_t o_h2   = o_y2 + (size_t)Nn * Hh;
  const size_t o_ns1  = o_h2 + (size_t)Nn * Hh;
  const size_t o_ns2  = o_ns1 + (size_t)Nn * Hh;
  const size_t o_pool = o_ns2 + (size_t)Nn * Hh;
  const size_t o_a1   = o_pool + (size_t)Bb * Hh;
  const size_t o_a2   = o_a1 + (size_t)Bb * Hh;
  const size_t o_pack = o_a2 + (size_t)Bb * Hh;             // bf16 region starts here

  float* agg1 = ws + o_agg1;  float* agg2 = ws + o_agg2;
  float* sums1 = ws + o_stat;       float* sq1 = ws + o_stat + 256;
  float* sums2 = ws + o_stat + 512; float* sq2 = ws + o_stat + 768;
  float* y1 = ws + o_y1;  float* h1 = ws + o_h1;
  float* y2 = ws + o_y2;  float* h2 = ws + o_h2;
  float* ns1 = ws + o_ns1; float* ns2 = ws + o_ns2;
  float* pool = ws + o_pool; float* a1 = ws + o_a1; float* a2 = ws + o_a2;

  unsigned short* pk = (unsigned short*)(ws + o_pack);
  unsigned short* pWe1 = pk;               // 32*128
  unsigned short* pWe2 = pWe1 + 4096;      // 32*256
  unsigned short* pW1  = pWe2 + 8192;      // 128*256
  unsigned short* pW2  = pW1 + 32768;      // 256*256
  unsigned short* pWn1 = pW2 + 65536;
  unsigned short* pWn2 = pWn1 + 65536;

  // ---- 0) zero agg1, agg2, BN stats (one contiguous prefix) ----
  k_zero<<<2048, 256, 0, stream>>>(ws, (int)o_y1);

  // ---- 1) pack weights into WMMA-B bf16 layout ----
  k_pack<<<1,  256, 0, stream>>>(We1, pWe1, EDd, Ff);
  k_pack<<<2,  256, 0, stream>>>(We2, pWe2, EDd, Hh);
  k_pack<<<8,  256, 0, stream>>>(W1,  pW1,  Ff,  Hh);
  k_pack<<<16, 256, 0, stream>>>(W2,  pW2,  Hh,  Hh);
  k_pack<<<16, 256, 0, stream>>>(Wn1, pWn1, Hh,  Hh);
  k_pack<<<16, 256, 0, stream>>>(Wn2, pWn2, Hh,  Hh);

  const int edgeBlocks = Ee / 128;                   // 16 edges/wave, 8 waves/block
  const int gemmBlocks = (Nn / 128) * (Hh / 64);     // m-blocks x n-slices

  // ---- conv1 ----
  k_edge<<<edgeBlocks, 256, 0, stream>>>(ea, pWe1, be1, x, agg1, srcI, dstI, Ff);
  k_gemm<<<gemmBlocks, 256, 0, stream>>>(x, agg1, pW1, b1, y1, Nn, Ff, Hh, 0);
  k_colstats<<<256, Hh, 0, stream>>>(y1, sums1, sq1, Nn / 256);
  k_bnapply<<<2048, 256, 0, stream>>>(y1, h1, sums1, sq1, g1, bt1, Nn * Hh, 1, 1.f / Nn);

  // ---- conv2 ----
  k_edge<<<edgeBlocks, 256, 0, stream>>>(ea, pWe2, be2, h1, agg2, srcI, dstI, Hh);
  k_gemm<<<gemmBlocks, 256, 0, stream>>>(h1, agg2, pW2, b2, y2, Nn, Hh, Hh, 0);
  k_colstats<<<256, Hh, 0, stream>>>(y2, sums2, sq2, Nn / 256);
  k_bnapply<<<2048, 256, 0, stream>>>(y2, h2, sums2, sq2, g2, bt2, Nn * Hh, 2, 1.f / Nn);

  // ---- node MLP ----
  k_gemm<<<gemmBlocks, 256, 0, stream>>>(h2,  nullptr, pWn1, bn1, ns1, Nn, Hh, Hh, 2);
  k_gemm<<<gemmBlocks, 256, 0, stream>>>(ns1, nullptr, pWn2, bn2, ns2, Nn, Hh, Hh, 2);
  k_nodescore<<<Nn / 8, 256, 0, stream>>>(ns2, Wn3, bn3, out + Bb * Aa);

  // ---- action head ----
  k_pool<<<Bb, Hh, 0, stream>>>(h2, pool);
  k_rowgemm<<<Bb, Hh, 0, stream>>>(pool, Wa1, ba1, a1);
  k_smallbn<<<1, Hh, 0, stream>>>(a1, ga1, bta1);
  k_rowgemm<<<Bb, Hh, 0, stream>>>(a1, Wa2, ba2, a2);
  k_smallbn<<<1, Hh, 0, stream>>>(a2, ga2, bta2);
  k_actionout<<<1, 640, 0, stream>>>(a2, Wa3, ba3, out);
}
